// LSTMClassifier_50766513439125
// MI455X (gfx1250) — matrix operand order repaired and checked by reading, not compiled, not run on hardware
//
#include <hip/hip_runtime.h>
#include <hip/hip_bf16.h>

#define T_STEPS 2048
#define BATCH   256
#define IN_DIM  6
#define HID     256
#define FC_DIM  128
#define OUT_DIM 10
#define NWG_TOTAL 32
#define THREADS 256

typedef __bf16 bf16_t;
typedef __attribute__((ext_vector_type(16))) __bf16 v16bf;
typedef __attribute__((ext_vector_type(8)))  __bf16 v8bf;
typedef __attribute__((ext_vector_type(8)))  float  v8f;

union BF16x16 {
    v16bf v;
    struct { v8bf lo, hi; } p;
};

// A fragment (16x32 bf16, M x K): lanes 0-15 -> rows, half picks K {8h..8h+7, 16+8h..16+8h+7}
__device__ __forceinline__ v16bf load_a_frag(const bf16_t* __restrict__ tile, int lane) {
    int row  = lane & 15;
    int half = lane >> 4;
    const bf16_t* p = tile + row * HID + half * 8;
    BF16x16 u;
    u.p.lo = *(const v8bf*)(p);
    u.p.hi = *(const v8bf*)(p + 16);
    return u.v;
}

// B fragment (32x16 bf16, K x N) from row-major W[N][K] slice:
// lane n = N column, half picks K {8h..8h+7, 16+8h..16+8h+7}
__device__ __forceinline__ v16bf load_b_frag(const bf16_t* __restrict__ wbase, int lane) {
    int n    = lane & 15;
    int half = lane >> 4;
    const bf16_t* p = wbase + n * HID + half * 8;
    BF16x16 u;
    u.p.lo = *(const v8bf*)(p);
    u.p.hi = *(const v8bf*)(p + 16);
    return u.v;
}

__device__ __forceinline__ float sigm(float x) { return 1.0f / (1.0f + __expf(-x)); }
__device__ __forceinline__ float tanh_f(float x) {
    float ax = fabsf(x);
    float e  = __expf(2.0f * ax);
    float t  = 1.0f - 2.0f / (e + 1.0f);
    return copysignf(t, x);
}

// Grid-wide generation barrier: bar[0]=count, bar[1]=generation
__device__ __forceinline__ void grid_sync(int* bar, int nblocks) {
    __syncthreads();
    if (threadIdx.x == 0) {
        __threadfence();
        volatile int* vgen = bar + 1;
        int g   = *vgen;
        int old = atomicAdd(bar, 1);
        if (old == nblocks - 1) {
            bar[0] = 0;
            __threadfence();
            atomicAdd(bar + 1, 1);
        } else {
            while (*vgen == g) __builtin_amdgcn_s_sleep(2);
            __threadfence();
        }
    }
    __syncthreads();
}

__global__ void __launch_bounds__(THREADS)
lstm_persistent(const float* __restrict__ x,
                const float* __restrict__ wih0,
                const float* __restrict__ bih0, const float* __restrict__ bhh0,
                const float* __restrict__ bih1, const float* __restrict__ bhh1,
                const bf16_t* __restrict__ whh0_bf,
                const bf16_t* __restrict__ wih1_bf,
                const bf16_t* __restrict__ whh1_bf,
                bf16_t* __restrict__ h0buf, bf16_t* __restrict__ h1buf,
                float* __restrict__ h0fin, float* __restrict__ h1fin,
                int* __restrict__ bar)
{
    // 64 KB LDS total. s_wb doubles as {x[t], Wih0 slice} for layer-0 WGs.
    __shared__ __attribute__((aligned(16))) bf16_t s_wa[64 * HID];   // Whh slice  (32 KB)
    __shared__ __attribute__((aligned(16))) bf16_t s_wb[64 * HID];   // Wih1 slice (32 KB) / layer0 scratch

    float* s_x    = reinterpret_cast<float*>(s_wb);                  // [256*6] floats
    float* s_wih0 = reinterpret_cast<float*>(s_wb) + BATCH * IN_DIM; // [64*6]  floats

    const int tid   = threadIdx.x;
    const int lane  = tid & 31;
    const int wv    = tid >> 5;       // wave id 0..7: owns M-tiles {2wv, 2wv+1}
    const int nloc  = lane & 15;
    const int half  = lane >> 4;
    const int layer = blockIdx.x >> 4;   // 0 or 1
    const int wg    = blockIdx.x & 15;   // hidden-chunk id
    const int j0    = wg * 16;           // hidden columns [j0, j0+16)

    // ---- one-time weight staging -------------------------------------------
    for (int i = tid; i < 64 * HID; i += THREADS) {
        int r = i >> 8;                              // local row 0..63 (gate*16 + n)
        int c = i & 255;
        int grow = (r >> 4) * HID + j0 + (r & 15);   // global gate row in [0,1024)
        if (layer == 0) {
            s_wa[i] = whh0_bf[(size_t)grow * HID + c];
        } else {
            s_wa[i] = whh1_bf[(size_t)grow * HID + c];
            s_wb[i] = wih1_bf[(size_t)grow * HID + c];
        }
    }
    if (layer == 0) {
        for (int i = tid; i < 64 * IN_DIM; i += THREADS) {
            int r = i / IN_DIM, c = i % IN_DIM;
            int grow = (r >> 4) * HID + j0 + (r & 15);
            s_wih0[i] = wih0[(size_t)grow * IN_DIM + c];
        }
    }
    // per-thread fused bias for the 4 gate columns this lane owns
    float bsum[4];
#pragma unroll
    for (int g = 0; g < 4; ++g) {
        int grow = g * HID + j0 + nloc;
        bsum[g] = layer ? (bih1[grow] + bhh1[grow]) : (bih0[grow] + bhh0[grow]);
    }
    __syncthreads();

    // c-state lives in registers for the whole sequence
    float creg[2][8];
#pragma unroll
    for (int mi = 0; mi < 2; ++mi)
#pragma unroll
        for (int j = 0; j < 8; ++j) creg[mi][j] = 0.0f;

    // ---- pipelined tick loop: layer0 does step k, layer1 does step k-1 -----
    for (int k = 0; k <= T_STEPS; ++k) {
        const bool act = (layer == 0) ? (k < T_STEPS) : (k >= 1);
        const int  t   = (layer == 0) ? k : (k - 1);

        if (layer == 0 && act) {
            const float* xt = x + (size_t)t * BATCH * IN_DIM;
            for (int i = tid; i < BATCH * IN_DIM; i += THREADS) s_x[i] = xt[i];
        }
        __syncthreads();

        if (act) {
            const size_t rdoff = (size_t)((k + 1) & 1) * BATCH * HID;
            const size_t wroff = (size_t)(k & 1) * BATCH * HID;
            const bf16_t* hprev = (layer ? h1buf : h0buf) + rdoff;
            bf16_t*       hout  = (layer ? h1buf : h0buf) + wroff;

            v8f acc[2][4];
#pragma unroll
            for (int g = 0; g < 4; ++g)
#pragma unroll
                for (int mi = 0; mi < 2; ++mi)
#pragma unroll
                    for (int j = 0; j < 8; ++j) acc[mi][g][j] = bsum[g];

            // recurrent GEMM: gates += h_prev @ Whh_slice^T   (K = 256, bf16 WMMA)
#pragma unroll
            for (int kb = 0; kb < 8; ++kb) {
                v16bf bw[4];
#pragma unroll
                for (int g = 0; g < 4; ++g)
                    bw[g] = load_b_frag(&s_wa[(g * 16) * HID + kb * 32], lane);
#pragma unroll
                for (int mi = 0; mi < 2; ++mi) {
                    const bf16_t* atile = hprev + (size_t)(2 * wv + mi) * 16 * HID + kb * 32;
                    v16bf a = load_a_frag(atile, lane);
#pragma unroll
                    for (int g = 0; g < 4; ++g)
                        acc[mi][g] = __builtin_amdgcn_wmma_f32_16x16x32_bf16(
                            false, a, false, bw[g], (short)0, acc[mi][g], false, false);
                }
            }

            if (layer == 1) {
                // input GEMM: gates += y0[t] @ Wih1_slice^T
                const bf16_t* y0 = h0buf + (size_t)((k - 1) & 1) * BATCH * HID;
#pragma unroll
                for (int kb = 0; kb < 8; ++kb) {
                    v16bf bw[4];
#pragma unroll
                    for (int g = 0; g < 4; ++g)
                        bw[g] = load_b_frag(&s_wb[(g * 16) * HID + kb * 32], lane);
#pragma unroll
                    for (int mi = 0; mi < 2; ++mi) {
                        const bf16_t* atile = y0 + (size_t)(2 * wv + mi) * 16 * HID + kb * 32;
                        v16bf a = load_a_frag(atile, lane);
#pragma unroll
                        for (int g = 0; g < 4; ++g)
                            acc[mi][g] = __builtin_amdgcn_wmma_f32_16x16x32_bf16(
                                false, a, false, bw[g], (short)0, acc[mi][g], false, false);
                    }
                }
            } else {
                // layer-0 input projection, K = 6: scalar FMAs from LDS
#pragma unroll
                for (int mi = 0; mi < 2; ++mi) {
                    int mt = 2 * wv + mi;
#pragma unroll
                    for (int g = 0; g < 4; ++g) {
                        const float* wrow = &s_wih0[(g * 16 + nloc) * IN_DIM];
#pragma unroll
                        for (int j = 0; j < 8; ++j) {
                            int b = mt * 16 + half * 8 + j;
                            const float* xb = &s_x[b * IN_DIM];
                            float s = 0.0f;
#pragma unroll
                            for (int kk = 0; kk < IN_DIM; ++kk) s += xb[kk] * wrow[kk];
                            acc[mi][g][j] += s;
                        }
                    }
                }
            }

            // elementwise LSTM cell update; D layout: (m = mt*16 + half*8 + j, n = nloc)
            const bool last = (layer == 0) ? (k == T_STEPS - 1) : (k == T_STEPS);
            float* hfin = layer ? h1fin : h0fin;
#pragma unroll
            for (int mi = 0; mi < 2; ++mi) {
                int mt = 2 * wv + mi;
#pragma unroll
                for (int j = 0; j < 8; ++j) {
                    float iv = acc[mi][0][j], fv = acc[mi][1][j];
                    float gv = acc[mi][2][j], ov = acc[mi][3][j];
                    float c  = sigm(fv) * creg[mi][j] + sigm(iv) * tanh_f(gv);
                    creg[mi][j] = c;
                    float h  = sigm(ov) * tanh_f(c);
                    int b    = mt * 16 + half * 8 + j;
                    int col  = j0 + nloc;
                    hout[(size_t)b * HID + col] = (bf16_t)h;
                    if (last) hfin[(size_t)b * HID + col] = h;
                }
            }
        }

        grid_sync(bar, NWG_TOTAL);
    }
}

__global__ void cvt_bf16_kernel(const float* __restrict__ in, bf16_t* __restrict__ out, int n) {
    int i = blockIdx.x * blockDim.x + threadIdx.x;
    if (i < n) out[i] = (bf16_t)in[i];
}

__global__ void zero_kernel(unsigned int* __restrict__ p, int n) {
    int i = blockIdx.x * blockDim.x + threadIdx.x;
    if (i < n) p[i] = 0u;
}

__global__ void fc1_kernel(const float* __restrict__ h0fin, const float* __restrict__ h1fin,
                           const float* __restrict__ w, const float* __restrict__ b,
                           float* __restrict__ out) {
    int idx = blockIdx.x * blockDim.x + threadIdx.x;   // [0, BATCH*FC_DIM)
    int bi = idx >> 7, j = idx & 127;
    const float* wr = w + (size_t)j * (2 * HID);
    const float* h0 = h0fin + (size_t)bi * HID;
    const float* h1 = h1fin + (size_t)bi * HID;
    float s = b[j];
    for (int k2 = 0; k2 < HID; ++k2) s += h0[k2] * wr[k2];
    for (int k2 = 0; k2 < HID; ++k2) s += h1[k2] * wr[HID + k2];
    out[idx] = s;
}

__global__ void fc2_kernel(const float* __restrict__ mid,
                           const float* __restrict__ w, const float* __restrict__ b,
                           float* __restrict__ out) {
    int idx = blockIdx.x * blockDim.x + threadIdx.x;
    if (idx >= BATCH * OUT_DIM) return;
    int bi = idx / OUT_DIM, o = idx % OUT_DIM;
    const float* m  = mid + (size_t)bi * FC_DIM;
    const float* wr = w + (size_t)o * FC_DIM;
    float s = b[o];
    for (int k2 = 0; k2 < FC_DIM; ++k2) s += m[k2] * wr[k2];
    out[idx] = s;
}

extern "C" void kernel_launch(void* const* d_in, const int* in_sizes, int n_in,
                              void* d_out, int out_size, void* d_ws, size_t ws_size,
                              hipStream_t stream) {
    (void)in_sizes; (void)n_in; (void)out_size; (void)ws_size;
    const float* x     = (const float*)d_in[0];
    const float* W_ih0 = (const float*)d_in[1];
    const float* W_hh0 = (const float*)d_in[2];
    const float* b_ih0 = (const float*)d_in[3];
    const float* b_hh0 = (const float*)d_in[4];
    const float* W_ih1 = (const float*)d_in[5];
    const float* W_hh1 = (const float*)d_in[6];
    const float* b_ih1 = (const float*)d_in[7];
    const float* b_hh1 = (const float*)d_in[8];
    const float* fc1_w = (const float*)d_in[9];
    const float* fc1_b = (const float*)d_in[10];
    const float* fc2_w = (const float*)d_in[11];
    const float* fc2_b = (const float*)d_in[12];

    // workspace layout (zero-region first: h0buf, h1buf, barrier)
    char* ws = (char*)d_ws;
    bf16_t* h0buf   = (bf16_t*)ws; ws += 2 * BATCH * HID * sizeof(bf16_t);   // 256 KB
    bf16_t* h1buf   = (bf16_t*)ws; ws += 2 * BATCH * HID * sizeof(bf16_t);   // 256 KB
    int*    bar     = (int*)ws;    ws += 256;
    bf16_t* whh0_bf = (bf16_t*)ws; ws += 4 * HID * HID * sizeof(bf16_t);
    bf16_t* wih1_bf = (bf16_t*)ws; ws += 4 * HID * HID * sizeof(bf16_t);
    bf16_t* whh1_bf = (bf16_t*)ws; ws += 4 * HID * HID * sizeof(bf16_t);
    float*  h0fin   = (float*)ws;  ws += BATCH * HID * sizeof(float);
    float*  h1fin   = (float*)ws;  ws += BATCH * HID * sizeof(float);
    float*  fcmid   = (float*)ws;  ws += BATCH * FC_DIM * sizeof(float);

    // zero state + barrier (runs every launch -> deterministic, graph-safe)
    int zero_n = (int)((2 * BATCH * HID * sizeof(bf16_t) * 2 + 256) / 4);
    zero_kernel<<<(zero_n + 255) / 256, 256, 0, stream>>>((unsigned int*)d_ws, zero_n);

    // fp32 -> bf16 weight conversion (stays resident in L2)
    int wn = 4 * HID * HID;
    cvt_bf16_kernel<<<(wn + 255) / 256, 256, 0, stream>>>(W_hh0, whh0_bf, wn);
    cvt_bf16_kernel<<<(wn + 255) / 256, 256, 0, stream>>>(W_ih1, wih1_bf, wn);
    cvt_bf16_kernel<<<(wn + 255) / 256, 256, 0, stream>>>(W_hh1, whh1_bf, wn);

    // persistent pipelined 2-layer LSTM (32 co-resident WGs, 1 barrier/step)
    lstm_persistent<<<NWG_TOTAL, THREADS, 0, stream>>>(
        x, W_ih0, b_ih0, b_hh0, b_ih1, b_hh1,
        whh0_bf, wih1_bf, whh1_bf,
        h0buf, h1buf, h0fin, h1fin, bar);

    // FC head
    fc1_kernel<<<(BATCH * FC_DIM) / 256, 256, 0, stream>>>(h0fin, h1fin, fc1_w, fc1_b, fcmid);
    fc2_kernel<<<(BATCH * OUT_DIM + 255) / 256, 256, 0, stream>>>(fcmid, fc2_w, fc2_b, (float*)d_out);
}
